// CrossAttentionFusion_27934467293654
// MI455X (gfx1250) — matrix operand order repaired and checked by reading, not compiled, not run on hardware
//
#include <hip/hip_runtime.h>

// Problem constants
#define DIM    2048
#define BATCH  16384

// Tiling
#define MT     128     // C rows per workgroup
#define NT     128     // C cols per workgroup
#define KTILE  32      // K per bf16 WMMA step
#define KPAD   40      // 32 + 8 bf16 pad (TDM pad: 16B per 64B row) -> conflict-free ds_load_b128
#define NIT    (DIM / KTILE)

typedef __attribute__((ext_vector_type(16))) __bf16 v16bf;
typedef __attribute__((ext_vector_type(8)))  __bf16 v8bf;
typedef __attribute__((ext_vector_type(4)))  __bf16 v4bf;
typedef __attribute__((ext_vector_type(8)))  float  v8f;
typedef __attribute__((ext_vector_type(4)))  unsigned int v4u;
typedef __attribute__((ext_vector_type(8)))  int v8i;
typedef __attribute__((ext_vector_type(4)))  int v4i;

// Generic pointer to an LDS object: low 32 bits are the workgroup-relative LDS byte offset.
static __device__ __forceinline__ unsigned lds_off(const void* p) {
    return (unsigned)(unsigned long long)(uintptr_t)p;
}

// One 2D TDM tile load: KTILE bf16 elements x tile_rows rows, tensor row pitch = DIM elements.
// TDM pad feature appends 4 DWORDs (16B) after every 16 DWORDs (one 64B row) -> LDS pitch = KPAD bf16.
static __device__ __forceinline__ void tdm_load_tile(unsigned lds_byte,
                                                     const __bf16* gsrc,
                                                     int tile_rows,
                                                     unsigned tensor_rows) {
    const unsigned long long ga = (unsigned long long)(uintptr_t)gsrc;
    v4u g0;
    g0[0] = 1u;                                                  // count=1, user descriptor
    g0[1] = lds_byte;                                            // lds_addr
    g0[2] = (unsigned)ga;                                        // global_addr[31:0]
    g0[3] = (unsigned)((ga >> 32) & 0x01ffffffu) | (2u << 30);   // global_addr[56:32] | type=2
    v8i g1;
    g1[0] = (int)((1u << 16)    // data_size = 2 bytes
                | (1u << 20)    // pad_enable
                | (3u << 22)    // pad_interval = 16 DWORDs (one row)
                | (3u << 25));  // pad_amount   = 4 DWORDs (16B)
    g1[1] = (int)(((unsigned)DIM & 0xffffu) << 16);              // tensor_dim0[15:0] (barrier addr = 0)
    g1[2] = (int)((((unsigned)DIM >> 16) & 0xffffu)
                | ((tensor_rows & 0xffffu) << 16));              // tensor_dim0[31:16] | tensor_dim1[15:0]
    g1[3] = (int)(((tensor_rows >> 16) & 0xffffu)
                | ((unsigned)KTILE << 16));                      // tensor_dim1[31:16] | tile_dim0
    g1[4] = tile_rows;                                           // tile_dim1 (tile_dim2 = 0)
    g1[5] = DIM;                                                 // tensor_dim0_stride[31:0]
    g1[6] = 0;
    g1[7] = 0;
    const v4i z = {0, 0, 0, 0};
#if __has_include(<hip/amd_detail/amd_gfx1250_TDM.h>)
    const v8i z8 = {0, 0, 0, 0, 0, 0, 0, 0};
    __builtin_amdgcn_tensor_load_to_lds(g0, g1, z, z, z8, 0);    // clang-23 / therock arity
#else
    __builtin_amdgcn_tensor_load_to_lds(g0, g1, z, z, 0);        // ROCm 7.2 clang-22 arity
#endif
}

// A-fragment (16x32, MxK), ISA 7.12.2:
//  lanes 0-15 : row M=lane,    K = {0..7, 16..23};  lanes 16-31: row M=lane-16, K = {8..15, 24..31}
static __device__ __forceinline__ v16bf frag_a(const __bf16 S[][KPAD], int baseRow, int lane) {
    const int half = lane >> 4;
    const int mr   = lane & 15;
    const v8bf c0 = *(const v8bf*)&S[baseRow + mr][half * 8];
    const v8bf c1 = *(const v8bf*)&S[baseRow + mr][16 + half * 8];
    return __builtin_shufflevector(c0, c1, 0,1,2,3,4,5,6,7,8,9,10,11,12,13,14,15);
}

// B-fragment (32x16, KxN), B staged as [n][k]:
//  lanes 0-15 : col N=lane, K = {0..7, 16..23};  lanes 16-31: col N=lane-16, K = {8..15, 24..31}
static __device__ __forceinline__ v16bf frag_b(const __bf16 S[][KPAD], int baseCol, int lane) {
    const int half = lane >> 4;
    const int n    = lane & 15;
    const v8bf c0 = *(const v8bf*)&S[baseCol + n][half * 8];
    const v8bf c1 = *(const v8bf*)&S[baseCol + n][16 + half * 8];
    return __builtin_shufflevector(c0, c1, 0,1,2,3,4,5,6,7,8,9,10,11,12,13,14,15);
}

// Elementwise fp32 -> bf16 (hi, lo) split: x = hi + lo with hi = bf16(x), lo = bf16(x - hi).
__global__ __launch_bounds__(256)
void split_bf16(const float* __restrict__ in, __bf16* __restrict__ hi, __bf16* __restrict__ lo) {
    const size_t i4 = ((size_t)blockIdx.x * 256 + threadIdx.x) * 4;
    const float4 v = *(const float4*)(in + i4);
    const float f[4] = {v.x, v.y, v.z, v.w};
    v4bf h, l;
    #pragma unroll
    for (int j = 0; j < 4; ++j) {
        h[j] = (__bf16)f[j];
        l[j] = (__bf16)(f[j] - (float)h[j]);
    }
    *(v4bf*)(hi + i4) = h;
    *(v4bf*)(lo + i4) = l;
}

// C = A @ W^T + bias (+ extra), bf16x3 (hi*hi + hi*lo + lo*hi) with fp32 accumulation.
// Tiles stream global -> LDS via TDM (tensor_load_to_lds), double-buffered, s_wait_tensorcnt synced.
// Per-wave tile: 32(M) x 64(N) = 2x4 WMMA subtiles -> 24 WMMA per 12 fragment loads per K-step.
template <bool CONVERT_A, bool HAS_EXTRA, bool OUT_SPLIT>
__global__ __launch_bounds__(256)
void gemm_bf16x3(const float*  __restrict__ Af,   // fp32 A (CONVERT_A only)
                 const __bf16* __restrict__ Ah,
                 const __bf16* __restrict__ Al,
                 const __bf16* __restrict__ Bh,
                 const __bf16* __restrict__ Bl,
                 const float*  __restrict__ bias,
                 const float*  __restrict__ extra,
                 float*        __restrict__ Cf,
                 __bf16*       __restrict__ Ch,
                 __bf16*       __restrict__ Cl)
{
    __shared__ __align__(16) __bf16 sAh[2][MT][KPAD];
    __shared__ __align__(16) __bf16 sAl[2][MT][KPAD];
    __shared__ __align__(16) __bf16 sBh[2][NT][KPAD];
    __shared__ __align__(16) __bf16 sBl[2][NT][KPAD];

    const int tid  = threadIdx.x;
    const int lane = tid & 31;
    const int w    = tid >> 5;   // 8 waves
    const int wm   = w & 3;      // 32-row group within 128-row tile
    const int wn   = w >> 2;     // 64-col group within 128-col tile
    const int row0 = blockIdx.y * MT;
    const int col0 = blockIdx.x * NT;
    constexpr int NTDM = CONVERT_A ? 2 : 4;   // TDM ops issued per K-step

    v8f acc[2][4];
    #pragma unroll
    for (int mi = 0; mi < 2; ++mi)
        #pragma unroll
        for (int ni = 0; ni < 4; ++ni)
            #pragma unroll
            for (int e = 0; e < 8; ++e)
                acc[mi][ni][e] = 0.0f;

    auto issue = [&](int buf, int k0) {
        if (!CONVERT_A) {
            tdm_load_tile(lds_off(&sAh[buf][0][0]), Ah + (size_t)row0 * DIM + k0, MT, BATCH);
            tdm_load_tile(lds_off(&sAl[buf][0][0]), Al + (size_t)row0 * DIM + k0, MT, BATCH);
        }
        tdm_load_tile(lds_off(&sBh[buf][0][0]), Bh + (size_t)col0 * DIM + k0, NT, DIM);
        tdm_load_tile(lds_off(&sBl[buf][0][0]), Bl + (size_t)col0 * DIM + k0, NT, DIM);
    };

    if (w == 0) issue(0, 0);  // preload tile 0

    for (int it = 0; it < NIT; ++it) {
        const int cur = it & 1;
        const int k0  = it * KTILE;

        // Prefetch next tile into the other buffer (freed by previous iteration's trailing barrier).
        if (w == 0 && it + 1 < NIT) issue(cur ^ 1, k0 + KTILE);

        if (CONVERT_A) {
            // Stage A tile fp32 -> bf16 hi/lo (overlaps with outstanding TDM B copies).
            #pragma unroll
            for (int i = 0; i < 4; ++i) {
                const int idx = tid + i * 256;
                const int r   = idx >> 3;
                const int c4  = (idx & 7) << 2;
                const float4 v = *(const float4*)(Af + (size_t)(row0 + r) * DIM + k0 + c4);
                const float f[4] = {v.x, v.y, v.z, v.w};
                #pragma unroll
                for (int j = 0; j < 4; ++j) {
                    const __bf16 h = (__bf16)f[j];
                    sAh[cur][r][c4 + j] = h;
                    sAl[cur][r][c4 + j] = (__bf16)(f[j] - (float)h);
                }
            }
        }

        // Current buffer's TDM copies complete when TENSORcnt drops to the newer batch's count.
        if (w == 0) {
            if (it + 1 < NIT) __builtin_amdgcn_s_wait_tensorcnt((short)NTDM);
            else              __builtin_amdgcn_s_wait_tensorcnt((short)0);
        }
        __syncthreads();

        v16bf bh[4], bl[4];
        #pragma unroll
        for (int ni = 0; ni < 4; ++ni) {
            bh[ni] = frag_b(sBh[cur], wn * 64 + ni * 16, lane);
            bl[ni] = frag_b(sBl[cur], wn * 64 + ni * 16, lane);
        }
        #pragma unroll
        for (int mi = 0; mi < 2; ++mi) {
            const v16bf ah = frag_a(sAh[cur], wm * 32 + mi * 16, lane);
            const v16bf al = frag_a(sAl[cur], wm * 32 + mi * 16, lane);
            #pragma unroll
            for (int ni = 0; ni < 4; ++ni) {
                acc[mi][ni] = __builtin_amdgcn_wmma_f32_16x16x32_bf16(
                    false, ah, false, bh[ni], (short)0, acc[mi][ni], false, false);
                acc[mi][ni] = __builtin_amdgcn_wmma_f32_16x16x32_bf16(
                    false, ah, false, bl[ni], (short)0, acc[mi][ni], false, false);
                acc[mi][ni] = __builtin_amdgcn_wmma_f32_16x16x32_bf16(
                    false, al, false, bh[ni], (short)0, acc[mi][ni], false, false);
            }
        }
        __syncthreads();  // all waves done reading buf[cur] before it is refilled
    }

    // Epilogue: C layout (ISA 7.12.2): VGPR r -> M = r + 8*(lane>=16), N = lane%16
    const int chalf = lane >> 4;
    const int cn    = lane & 15;
    #pragma unroll
    for (int mi = 0; mi < 2; ++mi) {
        #pragma unroll
        for (int ni = 0; ni < 4; ++ni) {
            const int rbase = row0 + wm * 32 + mi * 16 + chalf * 8;
            const int col   = col0 + wn * 64 + ni * 16 + cn;
            const float bcol = bias[col];
            #pragma unroll
            for (int r = 0; r < 8; ++r) {
                const size_t off = (size_t)(rbase + r) * DIM + col;
                float v = acc[mi][ni][r] + bcol;
                if (HAS_EXTRA) v += extra[off];
                if (OUT_SPLIT) {
                    const __bf16 h = (__bf16)v;
                    Ch[off] = h;
                    Cl[off] = (__bf16)(v - (float)h);
                } else {
                    Cf[off] = v;
                }
            }
        }
    }
}

extern "C" void kernel_launch(void* const* d_in, const int* in_sizes, int n_in,
                              void* d_out, int out_size, void* d_ws, size_t ws_size,
                              hipStream_t stream) {
    // setup_inputs order: emb_a, emb_b, Wq, bq, Wk, bk, Wv, bv, Wo, bo
    const float* emb_a = (const float*)d_in[0];
    const float* emb_b = (const float*)d_in[1];
    const float* Wv    = (const float*)d_in[6];
    const float* bv    = (const float*)d_in[7];
    const float* Wo    = (const float*)d_in[8];
    const float* bo    = (const float*)d_in[9];
    float* out = (float*)d_out;

    const size_t nX = (size_t)BATCH * DIM;   // 33,554,432 elems
    const size_t nW = (size_t)DIM * DIM;     //  4,194,304 elems

    // Workspace layout (bf16 planes)
    size_t off = 0;
    auto take = [&](size_t elems) { void* q = (char*)d_ws + off; off += elems * 2; return (__bf16*)q; };
    __bf16* Xh  = take(nX);
    __bf16* Xl  = take(nX);
    __bf16* Wvh = take(nW);
    __bf16* Wvl = take(nW);
    __bf16* Woh = take(nW);
    __bf16* Wol = take(nW);
    __bf16* Ebh = take(nX);
    __bf16* Ebl = take(nX);
    const bool pre_split_A = (ws_size >= off);

    dim3 block(256, 1, 1);
    dim3 grid(DIM / NT, BATCH / MT, 1);      // (16, 128)

    // Pre-split weights (and emb_b if workspace allows): one-time bandwidth-bound pass.
    split_bf16<<<dim3((unsigned)(nW / 1024)), block, 0, stream>>>(Wv, Wvh, Wvl);
    split_bf16<<<dim3((unsigned)(nW / 1024)), block, 0, stream>>>(Wo, Woh, Wol);

    // softmax over length-1 axis == 1 => out = (emb_b@Wv^T + bv + emb_a) @ Wo^T + bo
    if (pre_split_A) {
        split_bf16<<<dim3((unsigned)(nX / 1024)), block, 0, stream>>>(emb_b, Ebh, Ebl);
        gemm_bf16x3<false, true, true><<<grid, block, 0, stream>>>(
            nullptr, Ebh, Ebl, Wvh, Wvl, bv, emb_a, nullptr, Xh, Xl);
    } else {
        gemm_bf16x3<true, true, true><<<grid, block, 0, stream>>>(
            emb_b, nullptr, nullptr, Wvh, Wvl, bv, emb_a, nullptr, Xh, Xl);
    }
    gemm_bf16x3<false, false, false><<<grid, block, 0, stream>>>(
        nullptr, Xh, Xl, Woh, Wol, bo, nullptr, out, nullptr, nullptr);
}
